// MultiHeadAttention_84739704750464
// MI455X (gfx1250) — matrix operand order repaired and checked by reading, not compiled, not run on hardware
//
#include <hip/hip_runtime.h>
#include <hip/hip_bf16.h>

// MI455X (gfx1250) multi-head attention: bf16 WMMA (16x16x32), f32 accum,
// flash-style online softmax, async global->LDS staging where layout allows.

typedef __attribute__((ext_vector_type(16))) __bf16 v16bf;
typedef __attribute__((ext_vector_type(8)))  float  v8f;
typedef __attribute__((__vector_size__(4 * sizeof(int)))) int v4i_t;   // async-LDS builtin param type

union FragBF { v16bf v; uint4 q[2]; };

__device__ __forceinline__ unsigned short f2bf(float f) {
  unsigned int u = __float_as_uint(f);
  u += 0x7FFFu + ((u >> 16) & 1u);          // round-to-nearest-even
  return (unsigned short)(u >> 16);
}
__device__ __forceinline__ unsigned int pack2(float a, float b) {
  return (unsigned int)f2bf(a) | ((unsigned int)f2bf(b) << 16);
}

#define D_MODEL 1024
#define SEQ     2048
#define BATCH   2
#define HEADS   16
#define HDIM    64
#define MTOT    (BATCH * SEQ)   // 4096
#define LDSTR   40              // padded LDS row stride in ushorts

// ---- gfx1250 async global->LDS copy (ASYNCcnt) with graceful fallback -----
#if defined(__has_builtin)
#if __has_builtin(__builtin_amdgcn_global_load_async_to_lds_b128)
#define USE_ASYNC_LDS 1
#endif
#endif

__device__ __forceinline__ void wait_async0() {
#if defined(USE_ASYNC_LDS)
#if __has_builtin(__builtin_amdgcn_s_wait_asynccnt)
  __builtin_amdgcn_s_wait_asynccnt(0);
#else
  asm volatile("s_wait_asynccnt 0x0" ::: "memory");
#endif
#endif
}

// ---------------------------------------------------------------------------
// C[64x64] = A[64xK] * W[KxN block] + bias ; K-loop step 32, bf16 WMMA.
// OUT_MODE: 0 = f32 row-major, 1 = bf16 row-major, 2 = bf16 [b,h][d][s].
// ---------------------------------------------------------------------------
template<typename AT, int OUT_MODE>
__global__ __launch_bounds__(128)
void gemm_bias_kernel(const AT* __restrict__ A, const float* __restrict__ W,
                      const float* __restrict__ bias, void* __restrict__ Cout) {
  __shared__ unsigned short As[64 * LDSTR];
  __shared__ unsigned short Bs[64 * LDSTR];   // transposed: [n][k]

  const int t    = threadIdx.x;
  const int wave = t >> 5;
  const int lane = t & 31;
  const int r16  = lane & 15;
  const int hi   = lane >> 4;

  const int n0 = blockIdx.x * 64;
  const int m0 = blockIdx.y * 64;

  v8f acc[4];
  #pragma unroll
  for (int i = 0; i < 4; ++i)
    #pragma unroll
    for (int j = 0; j < 8; ++j) acc[i][j] = 0.f;

  for (int k0 = 0; k0 < D_MODEL; k0 += 32) {
    __syncthreads();
    // ---- stage A tile (64 rows x 32 k) -> LDS bf16 ----
    if constexpr (sizeof(AT) == 4) {
      const int row = t >> 1, half = t & 1;
      const float4* s = (const float4*)(A + (size_t)(m0 + row) * D_MODEL + k0 + half * 16);
      float4 f0 = s[0], f1 = s[1], f2 = s[2], f3 = s[3];
      unsigned int* d = (unsigned int*)&As[row * LDSTR + half * 16];
      d[0] = pack2(f0.x, f0.y); d[1] = pack2(f0.z, f0.w);
      d[2] = pack2(f1.x, f1.y); d[3] = pack2(f1.z, f1.w);
      d[4] = pack2(f2.x, f2.y); d[5] = pack2(f2.z, f2.w);
      d[6] = pack2(f3.x, f3.y); d[7] = pack2(f3.z, f3.w);
    } else {
      // bf16 A: rows are contiguous 64B in global -> async copy to LDS
#if defined(USE_ASYNC_LDS)
      #pragma unroll
      for (int j = 0; j < 2; ++j) {
        const int idx = t * 2 + j;
        const int row = idx >> 2, c = idx & 3;       // 4 x 16B chunks per row
        const AT* g = A + (size_t)(m0 + row) * D_MODEL + k0 + c * 8;
        __builtin_amdgcn_global_load_async_to_lds_b128(
            (v4i_t*)g, (v4i_t*)&As[row * LDSTR + c * 8], 0, 0);
      }
#else
      const int row = t >> 1, half = t & 1;
      const uint4* s = (const uint4*)(A + (size_t)(m0 + row) * D_MODEL + k0 + half * 16);
      uint4* d = (uint4*)&As[row * LDSTR + half * 16];
      d[0] = s[0]; d[1] = s[1];
#endif
    }
    // ---- stage B tile (32 k x 64 n) transposed -> Bs[n][k] ----
    {
      const int kr = t >> 2, nc = t & 3;
      const float4* s = (const float4*)(W + (size_t)(k0 + kr) * D_MODEL + n0 + nc * 16);
      float4 f0 = s[0], f1 = s[1], f2 = s[2], f3 = s[3];
      const float fv[16] = {f0.x, f0.y, f0.z, f0.w, f1.x, f1.y, f1.z, f1.w,
                            f2.x, f2.y, f2.z, f2.w, f3.x, f3.y, f3.z, f3.w};
      #pragma unroll
      for (int i = 0; i < 16; ++i)
        Bs[(nc * 16 + i) * LDSTR + kr] = f2bf(fv[i]);
      // prefetch next K-step of W through L2
      if (k0 + 32 < D_MODEL)
        __builtin_prefetch(W + (size_t)(k0 + 32 + kr) * D_MODEL + n0 + nc * 16, 0, 0);
    }
    if constexpr (sizeof(AT) == 2) wait_async0();
    __syncthreads();

    // B fragment: lane holds k = 8*hi..8*hi+7 and 16+8*hi..16+8*hi+7 of column (wave*16 + r16)
    FragBF bF;
    {
      const unsigned short* p = &Bs[(wave * 16 + r16) * LDSTR + hi * 8];
      bF.q[0] = *(const uint4*)(p);
      bF.q[1] = *(const uint4*)(p + 16);
    }
    #pragma unroll
    for (int mt = 0; mt < 4; ++mt) {
      FragBF aF;
      const unsigned short* p = &As[(mt * 16 + r16) * LDSTR + hi * 8];
      aF.q[0] = *(const uint4*)(p);
      aF.q[1] = *(const uint4*)(p + 16);
      acc[mt] = __builtin_amdgcn_wmma_f32_16x16x32_bf16(
          false, aF.v, false, bF.v, (short)0, acc[mt], false, false);
    }
  }

  const int n = n0 + wave * 16 + r16;
  const float bv = bias[n];
  #pragma unroll
  for (int mt = 0; mt < 4; ++mt) {
    #pragma unroll
    for (int r = 0; r < 8; ++r) {
      const int m = m0 + mt * 16 + r + 8 * hi;
      const float val = acc[mt][r] + bv;
      if constexpr (OUT_MODE == 0) {
        ((float*)Cout)[(size_t)m * D_MODEL + n] = val;
      } else if constexpr (OUT_MODE == 1) {
        ((unsigned short*)Cout)[(size_t)m * D_MODEL + n] = f2bf(val);
      } else {
        // V pre-transposed for attention: [b*H + h][d][s], bf16
        const int hh = n >> 6, dd = n & 63;
        const int bb = m >> 11, ss = m & (SEQ - 1);
        ((unsigned short*)Cout)[((size_t)(bb * HEADS + hh) * HDIM + dd) * SEQ + ss] = f2bf(val);
      }
    }
  }
}

// ---------------------------------------------------------------------------
// Flash attention: per (b,h): Q tile 64 rows, stream K/V in 32-key steps.
// V arrives pre-transposed [b,h][d][s] -> async straight copy into LDS.
// ---------------------------------------------------------------------------
__global__ __launch_bounds__(128)
void attention_kernel(const unsigned short* __restrict__ Qb,
                      const unsigned short* __restrict__ Kb,
                      const unsigned short* __restrict__ Vt,
                      unsigned short* __restrict__ Ob) {
  __shared__ unsigned short Vs[64 * LDSTR];   // [d][s-tile]
  __shared__ unsigned short Ps[64 * LDSTR];   // per-wave P tiles (C->A relayout)

  const int t = threadIdx.x;
  const int wave = t >> 5, lane = t & 31;
  const int r16 = lane & 15, hi = lane >> 4;

  const int bh = blockIdx.y;
  const int b = bh >> 4, h = bh & 15;
  const int q0 = blockIdx.x * 64;
  const size_t rowBase = (size_t)b * SEQ;
  const size_t vBase   = (size_t)bh * HDIM * SEQ;   // Vt[b,h] block: 64 rows x SEQ

  // Preload Q A-fragments (16 rows x head-dim 64 = two 16x32 chunks)
  FragBF aQ[2];
  {
    const int qrow = q0 + wave * 16 + r16;
    const unsigned short* qp = Qb + (rowBase + qrow) * D_MODEL + h * HDIM;
    #pragma unroll
    for (int c = 0; c < 2; ++c) {
      const unsigned short* p = qp + c * 32 + hi * 8;
      aQ[c].q[0] = *(const uint4*)(p);
      aQ[c].q[1] = *(const uint4*)(p + 16);
    }
  }

  float mrow[8], lrow[8];
  #pragma unroll
  for (int r = 0; r < 8; ++r) { mrow[r] = -1e30f; lrow[r] = 0.f; }
  v8f oAcc[4];
  #pragma unroll
  for (int i = 0; i < 4; ++i)
    #pragma unroll
    for (int j = 0; j < 8; ++j) oAcc[i][j] = 0.f;

  for (int s0 = 0; s0 < SEQ; s0 += 32) {
    __syncthreads();
    // ---- stage V tile: 64 d-rows x 32 keys, straight copy (pre-transposed)
#if defined(USE_ASYNC_LDS)
    #pragma unroll
    for (int j = 0; j < 2; ++j) {
      const int idx = t * 2 + j;
      const int row = idx >> 2, c = idx & 3;       // 4 x 16B chunks per d-row
      const unsigned short* g = Vt + vBase + (size_t)row * SEQ + s0 + c * 8;
      __builtin_amdgcn_global_load_async_to_lds_b128(
          (v4i_t*)g, (v4i_t*)&Vs[row * LDSTR + c * 8], 0, 0);
    }
    wait_async0();
#else
    {
      const int row = t >> 1, half = t & 1;
      const uint4* s = (const uint4*)(Vt + vBase + (size_t)row * SEQ + s0 + half * 16);
      uint4* d = (uint4*)&Vs[row * LDSTR + half * 16];
      d[0] = s[0]; d[1] = s[1];
    }
#endif
    __syncthreads();

    // ---- scores S = Q @ K^T : two 16x16 tiles, K B-frags direct from global
    v8f sAcc[2];
    #pragma unroll
    for (int i = 0; i < 2; ++i)
      #pragma unroll
      for (int j = 0; j < 8; ++j) sAcc[i][j] = 0.f;
    #pragma unroll
    for (int c = 0; c < 2; ++c) {
      #pragma unroll
      for (int nt = 0; nt < 2; ++nt) {
        FragBF bK;  // B = K^T: column n is key row (s0+nt*16+r16), k = head dim
        const int srow = s0 + nt * 16 + r16;
        const unsigned short* p = Kb + (rowBase + srow) * D_MODEL + h * HDIM + c * 32 + hi * 8;
        bK.q[0] = *(const uint4*)(p);
        bK.q[1] = *(const uint4*)(p + 16);
        sAcc[nt] = __builtin_amdgcn_wmma_f32_16x16x32_bf16(
            false, aQ[c].v, false, bK.v, (short)0, sAcc[nt], false, false);
      }
    }

    // ---- online softmax over this 32-key slab ----
    const float sc = 0.125f;  // 1/sqrt(HDIM)
    #pragma unroll
    for (int r = 0; r < 8; ++r) {
      const float s0v = sAcc[0][r] * sc;
      const float s1v = sAcc[1][r] * sc;
      float mx = fmaxf(s0v, s1v);
      #pragma unroll
      for (int off = 1; off < 16; off <<= 1)
        mx = fmaxf(mx, __shfl_xor(mx, off, 32));
      const float mnew = fmaxf(mrow[r], mx);
      const float scale_old = __expf(mrow[r] - mnew);
      const float p0 = __expf(s0v - mnew);
      const float p1 = __expf(s1v - mnew);
      float sum = p0 + p1;
      #pragma unroll
      for (int off = 1; off < 16; off <<= 1)
        sum += __shfl_xor(sum, off, 32);
      lrow[r] = lrow[r] * scale_old + sum;
      mrow[r] = mnew;
      #pragma unroll
      for (int dt = 0; dt < 4; ++dt) oAcc[dt][r] *= scale_old;
      // spill P (C layout) to per-wave LDS tile for A-layout reload
      const int prow = wave * 16 + r + 8 * hi;
      Ps[prow * LDSTR + r16]      = f2bf(p0);
      Ps[prow * LDSTR + 16 + r16] = f2bf(p1);
    }

    // ---- O += P (16x32) @ V (32x64) ----
    FragBF aP;
    {
      const unsigned short* p = &Ps[(wave * 16 + r16) * LDSTR + hi * 8];
      aP.q[0] = *(const uint4*)(p);
      aP.q[1] = *(const uint4*)(p + 16);
    }
    #pragma unroll
    for (int dt = 0; dt < 4; ++dt) {
      FragBF bV;
      const unsigned short* p = &Vs[(dt * 16 + r16) * LDSTR + hi * 8];
      bV.q[0] = *(const uint4*)(p);
      bV.q[1] = *(const uint4*)(p + 16);
      oAcc[dt] = __builtin_amdgcn_wmma_f32_16x16x32_bf16(
          false, aP.v, false, bV.v, (short)0, oAcc[dt], false, false);
    }
  }

  // ---- normalize and store O as bf16 (feeds final projection GEMM) ----
  #pragma unroll
  for (int dt = 0; dt < 4; ++dt) {
    #pragma unroll
    for (int r = 0; r < 8; ++r) {
      const int qr = q0 + wave * 16 + r + 8 * hi;
      const int col = h * HDIM + dt * 16 + r16;
      Ob[(rowBase + qr) * D_MODEL + col] = f2bf(oAcc[dt][r] / lrow[r]);
    }
  }
}

// ---------------------------------------------------------------------------
extern "C" void kernel_launch(void* const* d_in, const int* in_sizes, int n_in,
                              void* d_out, int out_size, void* d_ws, size_t ws_size,
                              hipStream_t stream) {
  (void)in_sizes; (void)n_in; (void)out_size; (void)ws_size;
  const float* q   = (const float*)d_in[0];
  const float* k   = (const float*)d_in[1];
  const float* v   = (const float*)d_in[2];
  const float* w_q = (const float*)d_in[3];
  const float* b_q = (const float*)d_in[4];
  const float* w_k = (const float*)d_in[5];
  const float* b_k = (const float*)d_in[6];
  const float* w_v = (const float*)d_in[7];
  const float* b_v = (const float*)d_in[8];
  const float* w_o = (const float*)d_in[9];
  const float* b_o = (const float*)d_in[10];

  unsigned short* Qb = (unsigned short*)d_ws;            // 8 MB each
  unsigned short* Kb = Qb + (size_t)MTOT * D_MODEL;
  unsigned short* Vt = Kb + (size_t)MTOT * D_MODEL;      // [b,h][d][s]
  unsigned short* Ob = Vt + (size_t)MTOT * D_MODEL;      // total 32 MB

  dim3 gG(D_MODEL / 64, MTOT / 64);   // (16, 64)
  gemm_bias_kernel<float, 1><<<gG, 128, 0, stream>>>(q, w_q, b_q, Qb);
  gemm_bias_kernel<float, 1><<<gG, 128, 0, stream>>>(k, w_k, b_k, Kb);
  gemm_bias_kernel<float, 2><<<gG, 128, 0, stream>>>(v, w_v, b_v, Vt);

  attention_kernel<<<dim3(SEQ / 64, BATCH * HEADS), 128, 0, stream>>>(Qb, Kb, Vt, Ob);

  gemm_bias_kernel<unsigned short, 0><<<gG, 128, 0, stream>>>(Ob, w_o, b_o, d_out);
}
